// Readout_41162966565599
// MI455X (gfx1250) — matrix operand order repaired and checked by reading, not compiled, not run on hardware
//
#include <hip/hip_runtime.h>
#include <hip/hip_bf16.h>
#include <cstdint>

typedef __attribute__((ext_vector_type(16))) __bf16 v16bf;
typedef __attribute__((ext_vector_type(8)))  float  v8f;

#define N_NEURONS 8192
#define N_CONV    64
#define BATCH     256
#define LYD       36
#define LXD       36
#define YX        (LYD * LXD)      // 1296
#define STAGE_YX  4
#define NSTAGES   (YX / STAGE_YX)  // 324
#define MTILE     32               // i rows per wave (two 16-row WMMA tiles)
#define ROWU      72               // padded row stride in ushorts (144 B): conflict-free b128
#define SLOTU     (MTILE * ROWU)   // 2304 ushorts per yx slot (4608 B)

__device__ __forceinline__ unsigned short f2bf(float f) {
    unsigned int u = __float_as_uint(f);
    unsigned int r = u + 0x7FFFu + ((u >> 16) & 1u);   // round to nearest even
    return (unsigned short)(r >> 16);
}

// packed bf16 x2 multiply — V_PK_MUL_BF16 (VOP3P op 42)
__device__ __forceinline__ unsigned int pk_mul_bf16(unsigned int a, unsigned int b) {
    unsigned int d;
    asm("v_pk_mul_bf16 %0, %1, %2" : "=v"(d) : "v"(a), "v"(b));
    return d;
}

// 64 B of global -> LDS via the CDNA5 async DMA path (ASYNCcnt-tracked).
// INST_OFFSET applies to both the LDS and global addresses (ISA 08 §4.4).
__device__ __forceinline__ void async_copy64(unsigned int lds_off, const void* gaddr) {
    asm volatile(
        "global_load_async_to_lds_b128 %0, %1, off\n\t"
        "global_load_async_to_lds_b128 %0, %1, off offset:16\n\t"
        "global_load_async_to_lds_b128 %0, %1, off offset:32\n\t"
        "global_load_async_to_lds_b128 %0, %1, off offset:48"
        :: "v"(lds_off), "v"(gaddr) : "memory");
}

__device__ __forceinline__ void wait_async0() {
    asm volatile("s_wait_asynccnt 0x0" ::: "memory");
}

// ---------------------------------------------------------------------------
// Pass 1: conv [B][C][yx] f32  ->  convT [yx][i][c] bf16  (c contiguous)
// ---------------------------------------------------------------------------
__global__ __launch_bounds__(256) void conv_transpose_bf16(
        const float* __restrict__ conv, unsigned short* __restrict__ convT)
{
    __shared__ float tile[64 * 16];
    const int i   = blockIdx.x;        // batch
    const int yx0 = blockIdx.y * 16;   // 81 tiles of 16 sites
    const int tid = threadIdx.x;
    {
        const int t  = tid & 15;       // yx within tile
        const int c0 = tid >> 4;
        const float* src = conv + (size_t)i * N_CONV * YX + yx0 + t;
        #pragma unroll
        for (int j = 0; j < 4; ++j) {
            const int c = c0 + j * 16;
            tile[c * 16 + t] = src[(size_t)c * YX];
        }
    }
    __syncthreads();
    {
        const int t  = tid >> 4;        // yx within tile
        const int cc = (tid & 15) * 4;  // c start
        ushort4 o;
        o.x = f2bf(tile[(cc + 0) * 16 + t]);
        o.y = f2bf(tile[(cc + 1) * 16 + t]);
        o.z = f2bf(tile[(cc + 2) * 16 + t]);
        o.w = f2bf(tile[(cc + 3) * 16 + t]);
        *(ushort4*)(convT + (((size_t)(yx0 + t) * BATCH + i) * N_CONV + cc)) = o;
    }
}

// ---------------------------------------------------------------------------
// Pass 2: per spatial site yx, per wave (32i x 16n tile):
//   B'(c,n) = Wc[c,n] * (Wy[n,y]*Wx[n,x])   (bf16 scale, v_pk_mul_bf16)
//   acc[i,n] = WMMA(A_yx[i,c], B'(c,n), acc[i,n])   (f32 accumulate in-place)
// A tiles (shared by all 8 waves) stream in via global_load_async_to_lds,
// double-buffered; one barrier per 4-site stage.
// ---------------------------------------------------------------------------
__global__ __launch_bounds__(256) void readout_wmma(
        const unsigned short* __restrict__ convT,
        const float* __restrict__ Wc,
        const float* __restrict__ Wy,
        const float* __restrict__ Wx,
        const float* __restrict__ bias,
        float* __restrict__ out)
{
    __shared__ unsigned short sWy[LYD * 128];                    // bf16
    __shared__ unsigned short sWx[LXD * 128];                    // bf16
    __shared__ alignas(16) unsigned short sA[2][STAGE_YX * SLOTU]; // 2 x 18.4KB

    const int tid    = threadIdx.x;
    const int wave   = tid >> 5;       // wave32
    const int lane   = tid & 31;
    const int laneN  = lane & 15;      // column n within tile (C/D layout)
    const int laneHi = lane >> 4;
    const int n0blk  = blockIdx.x * 128;
    const int i0     = blockIdx.y * MTILE;
    const int nl     = wave * 16 + laneN;
    const int n      = n0blk + nl;

    // Stage Wy / Wx (bf16) for this block's 128 neurons: layout [y][n_local]
    for (int idx = tid; idx < LYD * 128; idx += 256) {
        const int q = idx & 127, y = idx >> 7;
        sWy[idx] = f2bf(Wy[(size_t)(n0blk + q) * LYD + y]);
        sWx[idx] = f2bf(Wx[(size_t)(n0blk + q) * LXD + y]);
    }

    // B fragments: Wc[n, c] bf16, two K=32 chunks (c 0..31 / 32..63).
    unsigned int b0u[8], b1u[8];
    {
        const float* wp = Wc + (size_t)n * N_CONV + laneHi * 16;
        #pragma unroll
        for (int j = 0; j < 8; ++j) {
            b0u[j] = (unsigned int)f2bf(wp[2 * j]) |
                     ((unsigned int)f2bf(wp[2 * j + 1]) << 16);
            b1u[j] = (unsigned int)f2bf(wp[2 * j + 32]) |
                     ((unsigned int)f2bf(wp[2 * j + 33]) << 16);
        }
    }

    v8f acc0 = {0.f, 0.f, 0.f, 0.f, 0.f, 0.f, 0.f, 0.f};
    v8f acc1 = {0.f, 0.f, 0.f, 0.f, 0.f, 0.f, 0.f, 0.f};

    // Async staging map: 64 threads per yx slot; each thread owns 64 B of one
    // 128-B source i-row (so LDS/global deltas match despite LDS row padding).
    const int s    = tid >> 6;          // yx slot 0..3
    const int part = tid & 63;
    const int rowi = part >> 1;         // i row 0..31
    const int half = (part & 1) * 64;   // byte offset within row
    const unsigned int ldsA0 = (unsigned int)(size_t)&sA[0][0];
    const unsigned int myLds0 = ldsA0 + (unsigned int)(s * SLOTU * 2 + rowi * 144 + half);
    const unsigned int bufStride = STAGE_YX * SLOTU * 2;   // bytes between buffers

    { // prologue: stage 0
        const char* gp = (const char*)convT +
            (((size_t)s * BATCH + i0 + rowi) << 7) + half;   // (yx*256+i0+row)*128
        async_copy64(myLds0, gp);
        wait_async0();
    }
    __syncthreads();

    int buf = 0;
    for (int st = 0; st < NSTAGES; ++st) {
        if (st + 1 < NSTAGES) {   // issue next stage into back buffer
            const size_t yx = (size_t)(st + 1) * STAGE_YX + s;
            const char* gp = (const char*)convT +
                ((yx * BATCH + i0 + rowi) << 7) + half;
            async_copy64(myLds0 + (buf ^ 1) * bufStride, gp);
        }
        if (st + 2 < NSTAGES) {   // L2 prefetch two stages ahead
            __builtin_prefetch((const char*)convT +
                ((((size_t)(st + 2) * STAGE_YX + s) * BATCH + i0 + rowi) << 7) + half,
                0, 0);
        }
        #pragma unroll
        for (int ss = 0; ss < STAGE_YX; ++ss) {
            const int yx = st * STAGE_YX + ss;
            const int y  = yx / LXD;
            const int x  = yx - y * LXD;
            const unsigned int wy = sWy[y * 128 + nl];
            const unsigned int wx = sWx[x * 128 + nl];
            const unsigned int wyx2 = pk_mul_bf16(wy * 0x10001u, wx * 0x10001u);

            union BU { unsigned int u[8]; v16bf v; };
            BU bs0, bs1;
            #pragma unroll
            for (int j = 0; j < 8; ++j) {
                bs0.u[j] = pk_mul_bf16(b0u[j], wyx2);
                bs1.u[j] = pk_mul_bf16(b1u[j], wyx2);
            }

            union AV { uint4 u[2]; v16bf v; };
            AV a0, a1, a2, a3;
            const uint4* ap =
                (const uint4*)&sA[buf][ss * SLOTU + laneN * ROWU + laneHi * 16];
            a0.u[0] = ap[0];   a0.u[1] = ap[1];    // i 0..15,  c chunk0
            a1.u[0] = ap[4];   a1.u[1] = ap[5];    // i 0..15,  c chunk1
            a2.u[0] = ap[144]; a2.u[1] = ap[145];  // i 16..31 (+16*144B), chunk0
            a3.u[0] = ap[148]; a3.u[1] = ap[149];  // i 16..31, chunk1

            acc0 = __builtin_amdgcn_wmma_f32_16x16x32_bf16(false, a0.v, false, bs0.v,
                                                           (short)0, acc0, false, false);
            acc0 = __builtin_amdgcn_wmma_f32_16x16x32_bf16(false, a1.v, false, bs1.v,
                                                           (short)0, acc0, false, false);
            acc1 = __builtin_amdgcn_wmma_f32_16x16x32_bf16(false, a2.v, false, bs0.v,
                                                           (short)0, acc1, false, false);
            acc1 = __builtin_amdgcn_wmma_f32_16x16x32_bf16(false, a3.v, false, bs1.v,
                                                           (short)0, acc1, false, false);
        }
        wait_async0();       // our async loads into the back buffer are done
        __syncthreads();     // everyone's loads visible to all waves
        buf ^= 1;
    }

    // epilogue: bias + ELU, store [i, n] (f32), two 16-row tiles
    const float bn = bias[n];
    #pragma unroll
    for (int r = 0; r < 8; ++r) {
        const int row = laneHi * 8 + r;          // C/D layout: VGPR r -> row
        float v0 = acc0[r] + bn;
        float v1 = acc1[r] + bn;
        v0 = v0 > 0.0f ? v0 : (__expf(v0) - 1.0f);
        v1 = v1 > 0.0f ? v1 : (__expf(v1) - 1.0f);
        out[(size_t)(i0 + row) * N_NEURONS + n]      = v0;
        out[(size_t)(i0 + 16 + row) * N_NEURONS + n] = v1;
    }
}

// ---------------------------------------------------------------------------
extern "C" void kernel_launch(void* const* d_in, const int* in_sizes, int n_in,
                              void* d_out, int out_size, void* d_ws, size_t ws_size,
                              hipStream_t stream) {
    const float* conv = (const float*)d_in[0];
    const float* Wc   = (const float*)d_in[1];
    const float* Wy   = (const float*)d_in[2];
    const float* Wx   = (const float*)d_in[3];
    const float* bias = (const float*)d_in[4];
    float* out = (float*)d_out;
    unsigned short* convT = (unsigned short*)d_ws; // 1296*256*64 bf16 = 42.5 MB

    dim3 g1(BATCH, YX / 16);                 // 256 x 81
    conv_transpose_bf16<<<g1, 256, 0, stream>>>(conv, convT);

    dim3 g2(N_NEURONS / 128, BATCH / MTILE); // 64 n-groups x 8 i-tiles
    readout_wmma<<<g2, 256, 0, stream>>>(convT, Wc, Wy, Wx, bias, out);
}
